// PrunedCaps_4329327034760
// MI455X (gfx1250) — matrix operand order repaired and checked by reading, not compiled, not run on hardware
//
#include <hip/hip_runtime.h>

#define THETA 0.7f
#define EPSN  1e-8f

typedef __attribute__((ext_vector_type(16))) __bf16 v16bf;
typedef __attribute__((ext_vector_type(8)))  __bf16 v8bf;
typedef __attribute__((ext_vector_type(8)))  float  v8f;
typedef __attribute__((ext_vector_type(4)))  unsigned v4u;
typedef __attribute__((ext_vector_type(8)))  int    v8i;
typedef __attribute__((ext_vector_type(4)))  int    v4i;

static constexpr int NB = 16;    // batches
static constexpr int NC = 1024;  // capsules per batch
static constexpr int DD = 2048;  // flattened dim d*H*W

#if __has_builtin(__builtin_amdgcn_tensor_load_to_lds)
#define HAVE_TDM 1
#if __has_include(<hip/amd_detail/amd_gfx1250_TDM.h>)
#define TDM_6ARG 1
#endif
#endif

__device__ __forceinline__ unsigned short f2bf(float f) {
    unsigned u = __float_as_uint(f);
    u += 0x7FFFu + ((u >> 16) & 1u);   // round to nearest even
    return (unsigned short)(u >> 16);
}

// ---------------------------------------------------------------- kernel 1
// one wave32 per capsule: squared-norm reduce; also zero suppress flags
__global__ void __launch_bounds__(256)
pc_norms(const float* __restrict__ u, float* __restrict__ norms,
         float* __restrict__ inv, int* __restrict__ suppress) {
    int tid = blockIdx.x * 256 + threadIdx.x;
    if (tid < NB * NC) suppress[tid] = 0;
    int wave = threadIdx.x >> 5;
    int lane = threadIdx.x & 31;
    int row  = blockIdx.x * 8 + wave;               // capsule index (16384 total)
    const float4* r = (const float4*)(u + (size_t)row * DD);
    float s = 0.f;
#pragma unroll
    for (int i = 0; i < 16; ++i) {
        float4 v = r[lane + i * 32];
        s += v.x * v.x + v.y * v.y + v.z * v.z + v.w * v.w;
    }
#pragma unroll
    for (int off = 16; off >= 1; off >>= 1) s += __shfl_xor(s, off, 32);
    if (lane == 0) {
        float nrm = sqrtf(s);
        norms[row] = nrm;
        inv[row]   = 1.0f / fmaxf(nrm, EPSN);
    }
}

// ---------------------------------------------------------------- kernel 2
// per-batch bitonic sort (ascending by norm, idx tiebreak) in LDS
__global__ void __launch_bounds__(1024)
pc_sort(const float* __restrict__ norms, int* __restrict__ order) {
    __shared__ float key[NC];
    __shared__ int   idx[NC];
    const int b = blockIdx.x, t = threadIdx.x;
    key[t] = norms[b * NC + t];
    idx[t] = t;
    __syncthreads();
    for (int k = 2; k <= NC; k <<= 1) {
        for (int j = k >> 1; j > 0; j >>= 1) {
            int p = t ^ j;
            if (p > t) {
                bool up = ((t & k) == 0);
                float a = key[t], c2 = key[p];
                int   ia = idx[t], ic = idx[p];
                bool gt = (a > c2) || (a == c2 && ia > ic);
                bool sw = up ? gt : !gt && (a != c2 || ia != ic);
                if (sw) { key[t] = c2; key[p] = a; idx[t] = ic; idx[p] = ia; }
            }
            __syncthreads();
        }
    }
    order[b * NC + t] = idx[t];
}

// ---------------------------------------------------------------- kernel 3
// gather sorted rows, normalize, convert to bf16 (packed u16) in workspace
__global__ void __launch_bounds__(256)
pc_gather(const float* __restrict__ u, const int* __restrict__ order,
          const float* __restrict__ inv, unsigned short* __restrict__ unit) {
    int b = blockIdx.x >> 10;
    int p = blockIdx.x & (NC - 1);
    int src = order[b * NC + p];
    float s = inv[b * NC + src];
    const float4* srow = (const float4*)(u + (size_t)(b * NC + src) * DD);
    uint2* drow = (uint2*)(unit + (size_t)(b * NC + p) * DD);
#pragma unroll
    for (int i = 0; i < 2; ++i) {
        int e = threadIdx.x + i * 256;
        float4 v = srow[e];
        uint2 o;
        o.x = (unsigned)f2bf(v.x * s) | ((unsigned)f2bf(v.y * s) << 16);
        o.y = (unsigned)f2bf(v.z * s) | ((unsigned)f2bf(v.w * s) << 16);
        drow[e] = o;
    }
}

// ---------------------------------------------------------------- kernel 4
// Gram-matrix tiles via v_wmma_f32_16x16x32_bf16; upper-triangle suppression.
// Block = 8 waves; wave w owns a 16-row i-tile (block covers 128 rows).
// Per pass: 4 j-tiles (64 columns), A panel 128x64 staged with b128 copies,
// B panel 64x64 staged via the Tensor Data Mover (fallback: b128 copies).
__global__ void __launch_bounds__(256)
pc_suppress(const unsigned short* __restrict__ unit, int* __restrict__ suppress) {
    __shared__ __align__(32) unsigned short alds[128 * 64]; // A panel: 16 KB
    __shared__ __align__(32) unsigned short blds[64 * 64];  // B panel:  8 KB
    const int iblk = blockIdx.x;        // 0..7
    const int b    = blockIdx.y;        // 0..15
    const int t    = threadIdx.x;
    const int wave = t >> 5, lane = t & 31;
    const int half = lane >> 4, c = lane & 15;
    const int ibase = iblk * 128;
    const int i0    = ibase + wave * 16;
    const uint4* uni4 = (const uint4*)unit;                 // row = DD/8 uint4
    const size_t rb4 = (size_t)b * NC * (DD / 8);
    uint4* alds4 = (uint4*)alds;
    unsigned supMask = 0;

    for (int jt0 = iblk * 8; jt0 < NC / 16; jt0 += 4) {     // 4 j-tiles per pass
        v8f acc[4] = {};
        for (int kc = 0; kc < DD; kc += 64) {
            __syncthreads();
#ifdef HAVE_TDM
            if (wave == 0) {
                // TDM: 2-D tile (64 rows x 64 bf16) of row-major [NB*NC, DD] tensor -> LDS
                unsigned lds_b = (unsigned)(uintptr_t)(void*)blds;
                unsigned long long ga = (unsigned long long)(uintptr_t)unit +
                    ((unsigned long long)(b * NC + jt0 * 16) * DD + kc) * 2ull;
                v4u g0 = { 1u,                                   // count=1, no gather
                           lds_b,                                // lds_addr
                           (unsigned)ga,                         // global_addr[31:0]
                           ((unsigned)(ga >> 32) & 0x01FFFFFFu)  // global_addr[56:32]
                               | (2u << 30) };                   // type = 2 ("image")
                v8i g1 = { (int)0x00010000,        // wg_mask=0, data_size=1 (2 bytes)
                           (int)(DD << 16),        // tensor_dim0[15:0] = 2048
                           (int)((NB * NC) << 16), // tensor_dim1[15:0] = 16384
                           (int)(64 << 16),        // tile_dim0 = 64 (K elements)
                           64,                     // tile_dim1 = 64 (rows), tile_dim2 = 0
                           DD,                     // tensor_dim0_stride = 2048
                           0, 0 };
                v4i gz = { 0, 0, 0, 0 };
#ifdef TDM_6ARG
                v8i gz8 = { 0, 0, 0, 0, 0, 0, 0, 0 };
                __builtin_amdgcn_tensor_load_to_lds(g0, g1, gz, gz, gz8, 0);
#else
                __builtin_amdgcn_tensor_load_to_lds(g0, g1, gz, gz, 0);
#endif
            }
#endif
            // stage A: 128x64 bf16 = 1024 uint4, coalesced b128, 4 per thread
            const int k4 = kc >> 3;
#pragma unroll
            for (int it = 0; it < 4; ++it) {
                int q = t + it * 256;
                int row = q >> 3, c4 = q & 7;
                alds4[q] = uni4[rb4 + (size_t)(ibase + row) * (DD / 8) + k4 + c4];
            }
#ifndef HAVE_TDM
            // fallback B staging: 64x64 bf16 = 512 uint4, 2 per thread
            uint4* blds4 = (uint4*)blds;
#pragma unroll
            for (int it = 0; it < 2; ++it) {
                int q = t + it * 256;
                int row = q >> 3, c4 = q & 7;
                blds4[q] = uni4[rb4 + (size_t)(jt0 * 16 + row) * (DD / 8) + k4 + c4];
            }
#endif
            if (kc + 64 < DD)   // prefetch next A chunk (global_prefetch_b8)
                __builtin_prefetch(&uni4[rb4 + (size_t)(ibase + (t >> 1)) * (DD / 8) + k4 + 8], 0, 0);
#ifdef HAVE_TDM
            if (wave == 0) __builtin_amdgcn_s_wait_tensorcnt(0);
#endif
            __syncthreads();
            // 8 WMMAs per staging step: 2 K-halves x 4 independent accumulators
#pragma unroll
            for (int ko = 0; ko < 64; ko += 32) {
                // A (16x32 bf16): lane m=c holds row i0+m; K span contiguous
                // (k = half*16 + s), matching B.
                v8bf alo = *(const v8bf*)&alds[(wave * 16 + c) * 64 + ko + half * 16];
                v8bf ahi = *(const v8bf*)&alds[(wave * 16 + c) * 64 + ko + half * 16 + 8];
                v16bf afrag = __builtin_shufflevector(alo, ahi,
                    0, 1, 2, 3, 4, 5, 6, 7, 8, 9, 10, 11, 12, 13, 14, 15);
#pragma unroll
                for (int s = 0; s < 4; ++s) {
                    // B (32x16 bf16): lane holds column c of sub-tile s, K span contiguous
                    v16bf bfrag = *(const v16bf*)&blds[(s * 16 + c) * 64 + ko + half * 16];
                    acc[s] = __builtin_amdgcn_wmma_f32_16x16x32_bf16(
                        false, afrag, false, bfrag, (short)0, acc[s], false, false);
                }
            }
        }
        // C/D layout: VGPR r -> M = half*8 + r, N = c
#pragma unroll
        for (int s = 0; s < 4; ++s) {
            int j = (jt0 + s) * 16 + c;
#pragma unroll
            for (int r = 0; r < 8; ++r) {
                int i = i0 + half * 8 + r;
                if (j > i && acc[s][r] > THETA) supMask |= (1u << r);
            }
        }
    }
    // OR-reduce across the 16 lanes sharing the same half (rows identical there)
#pragma unroll
    for (int off = 1; off < 16; off <<= 1)
        supMask |= (unsigned)__shfl_xor((int)supMask, off, 32);
    if (c == 0) {
#pragma unroll
        for (int r = 0; r < 8; ++r)
            if (supMask & (1u << r)) suppress[b * NC + i0 + half * 8 + r] = 1;
    }
}

// ---------------------------------------------------------------- kernel 5
// out[p] = suppress[p] ? 0 : u[order[p]]  (unnormalized rows, per reference)
__global__ void __launch_bounds__(256)
pc_output(const float* __restrict__ u, const int* __restrict__ order,
          const int* __restrict__ suppress, float* __restrict__ out) {
    int b = blockIdx.x >> 10;
    int p = blockIdx.x & (NC - 1);
    int src = order[b * NC + p];
    int sup = suppress[b * NC + p];
    const float4* srow = (const float4*)(u + (size_t)(b * NC + src) * DD);
    float4* drow = (float4*)(out + (size_t)(b * NC + p) * DD);
    float4 z = {0.f, 0.f, 0.f, 0.f};
#pragma unroll
    for (int i = 0; i < 2; ++i) {
        int e = threadIdx.x + i * 256;
        drow[e] = sup ? z : srow[e];
    }
}

extern "C" void kernel_launch(void* const* d_in, const int* in_sizes, int n_in,
                              void* d_out, int out_size, void* d_ws, size_t ws_size,
                              hipStream_t stream) {
    const float* u = (const float*)d_in[0];
    float* out = (float*)d_out;
    char* ws = (char*)d_ws;
    // workspace layout (offsets 64KB-aligned): norms, inv, order, suppress, unit(bf16, 64 MiB)
    float* norms          = (float*)(ws);
    float* inv            = (float*)(ws + (1 << 16));
    int*   order          = (int*)  (ws + 2 * (1 << 16));
    int*   suppress       = (int*)  (ws + 3 * (1 << 16));
    unsigned short* unit  = (unsigned short*)(ws + 4 * (1 << 16));

    pc_norms   <<<dim3(2048),      dim3(256),  0, stream>>>(u, norms, inv, suppress);
    pc_sort    <<<dim3(NB),        dim3(1024), 0, stream>>>(norms, order);
    pc_gather  <<<dim3(NB * NC),   dim3(256),  0, stream>>>(u, order, inv, unit);
    pc_suppress<<<dim3(8, NB),     dim3(256),  0, stream>>>(unit, suppress);
    pc_output  <<<dim3(NB * NC),   dim3(256),  0, stream>>>(u, order, suppress, out);
}
